// MultiHeadAttention_83176336655034
// MI455X (gfx1250) — compile-verified, matrix operand order checked by reading, not run on hardware
//
#include <hip/hip_runtime.h>

#define S_  1024
#define B_  2
#define D_  1024
#define H_  16
#define HD_ 64
#define M_  (S_*B_)   // 2048 rows

typedef _Float16 v16h __attribute__((ext_vector_type(16)));
typedef _Float16 v8h  __attribute__((ext_vector_type(8)));
typedef float    v8f  __attribute__((ext_vector_type(8)));

static __device__ __forceinline__ v8f vzero8() {
    v8f z;
#pragma unroll
    for (int i = 0; i < 8; ++i) z[i] = 0.0f;
    return z;
}

// Load a 16x32 (f16) WMMA A/B operand chunk for this lane.
// p = start of this lane's row (A: row m; B: col n) at the current k-base.
// Lane group g (=lane>>4): g=0 takes halves {0..7,16..23}, g=1 takes {8..15,24..31}.
static __device__ __forceinline__ v16h ld_ab(const _Float16* p, int g) {
    v8h lo = *(const v8h*)(p + g * 8);
    v8h hi = *(const v8h*)(p + g * 8 + 16);
    return __builtin_shufflevector(lo, hi, 0,1,2,3,4,5,6,7,8,9,10,11,12,13,14,15);
}

static __device__ __forceinline__ v8f wmma16(v16h a, v16h b, v8f c) {
    // D = A(16x32,f16) * B(32x16,f16) + C(16x16,f32)
    return __builtin_amdgcn_wmma_f32_16x16x32_f16(false, a, false, b, (short)0, c, false, false);
}

// ---------------------------------------------------------------- convert
__global__ void cvt_f16_kernel(const float* __restrict__ src,
                               _Float16* __restrict__ dst, int n) {
    int i = blockIdx.x * blockDim.x + threadIdx.x;
    if (i < n) dst[i] = (_Float16)src[i];
}

// ---------------------------------------------------------------- QKV GEMM
// X (2048x1024, f16) @ W[which]^T + b, 64Mx64N per wave (4 M-tiles x 4 N-tiles):
// each A and B fetch feeds 4 WMMAs -> 1.0 b128 loads per WMMA in steady state.
// Outputs: xq/xk in (b,h,s,hd), xv transposed to (b,h,hd,t).
__global__ __launch_bounds__(32) void qkv_gemm_kernel(
    const _Float16* __restrict__ qf16, const _Float16* __restrict__ wf16,
    const float* __restrict__ bq, const float* __restrict__ bk,
    const float* __restrict__ bv,
    _Float16* __restrict__ xq, _Float16* __restrict__ xk,
    _Float16* __restrict__ xvT)
{
    const int lane   = threadIdx.x;
    const int colx   = lane & 15;
    const int g      = lane >> 4;
    const int mtile4 = blockIdx.x;  // 0..31 (64 rows each)
    const int h      = blockIdx.y;  // 0..15 (covers 64 output cols)
    const int which  = blockIdx.z;  // 0=q 1=k 2=v

    const _Float16* W    = wf16 + (size_t)which * D_ * D_;
    const float*    bias = (which == 0) ? bq : (which == 1) ? bk : bv;

    const _Float16* arow[4];
#pragma unroll
    for (int mt = 0; mt < 4; ++mt)
        arow[mt] = qf16 + (size_t)(mtile4 * 64 + mt * 16 + colx) * D_;

    v8f acc[4][4];
#pragma unroll
    for (int mt = 0; mt < 4; ++mt)
#pragma unroll
        for (int nt = 0; nt < 4; ++nt) acc[mt][nt] = vzero8();

    for (int k = 0; k < D_; k += 32) {
        v16h a[4];
#pragma unroll
        for (int mt = 0; mt < 4; ++mt) a[mt] = ld_ab(arow[mt] + k, g);
#pragma unroll
        for (int nt = 0; nt < 4; ++nt) {
            const _Float16* wrow = W + (size_t)(h * 64 + nt * 16 + colx) * D_;
            v16h bm = ld_ab(wrow + k, g);
#pragma unroll
            for (int mt = 0; mt < 4; ++mt)
                acc[mt][nt] = wmma16(a[mt], bm, acc[mt][nt]);
        }
    }

#pragma unroll
    for (int nt = 0; nt < 4; ++nt) {
        const int   n  = h * 64 + nt * 16 + colx;
        const int   hd = nt * 16 + colx;
        const float bn = bias[n];
#pragma unroll
        for (int mt = 0; mt < 4; ++mt) {
#pragma unroll
            for (int i = 0; i < 8; ++i) {
                const int m = mtile4 * 64 + mt * 16 + i + 8 * g;
                const int s = m >> 1;       // rows are (s,b) interleaved, B=2
                const int b = m & 1;
                const _Float16 val = (_Float16)(acc[mt][nt][i] + bn);
                if (which == 2) {
                    xvT[((size_t)((b * H_ + h) * HD_ + hd)) * S_ + s] = val;  // (b,h,hd,t)
                } else {
                    _Float16* dst = (which == 0) ? xq : xk;
                    dst[((size_t)((b * H_ + h) * S_ + s)) * HD_ + hd] = val;  // (b,h,s,hd)
                }
            }
        }
    }
}

// ---------------------------------------------------------------- flash attention
// One wave per (b,h,stile): 16 Q rows, streams all 1024 t columns 32 at a time.
__global__ __launch_bounds__(32) void attn_kernel(
    const _Float16* __restrict__ xq, const _Float16* __restrict__ xk,
    const _Float16* __restrict__ xvT, const float* __restrict__ mask,
    _Float16* __restrict__ attn)
{
    __shared__ _Float16 pLDS[16 * 32];   // 1 KB, wave-private (1 wave/WG)

    const int lane  = threadIdx.x;
    const int colx  = lane & 15;
    const int g     = lane >> 4;
    const int stile = blockIdx.x & 63;
    const int h     = (blockIdx.x >> 6) & 15;
    const int b     = blockIdx.x >> 10;

    const _Float16* Qb = xq  + ((size_t)(b * H_ + h) * S_) * HD_;
    const _Float16* Kb = xk  + ((size_t)(b * H_ + h) * S_) * HD_;
    const _Float16* Vt = xvT + ((size_t)(b * H_ + h) * HD_) * S_;

    // Q tile A-operands, kept resident: hd 0..31 and 32..63
    const _Float16* qrow = Qb + (size_t)(stile * 16 + colx) * HD_;
    const v16h qa0 = ld_ab(qrow,      g);
    const v16h qa1 = ld_ab(qrow + 32, g);

    float mrun[8], lrun[8];
#pragma unroll
    for (int i = 0; i < 8; ++i) { mrun[i] = -1.0e30f; lrun[i] = 0.0f; }
    v8f o[4];
#pragma unroll
    for (int nt = 0; nt < 4; ++nt) o[nt] = vzero8();

    for (int tb = 0; tb < S_; tb += 32) {
        // scores for 32 t columns: two 16x16 C tiles, K=64 over hd
        v8f s0 = vzero8(), s1 = vzero8();
        {
            const _Float16* krow0 = Kb + (size_t)(tb + colx) * HD_;
            s0 = wmma16(qa0, ld_ab(krow0,      g), s0);
            s0 = wmma16(qa1, ld_ab(krow0 + 32, g), s0);
            const _Float16* krow1 = Kb + (size_t)(tb + 16 + colx) * HD_;
            s1 = wmma16(qa0, ld_ab(krow1,      g), s1);
            s1 = wmma16(qa1, ld_ab(krow1 + 32, g), s1);
        }

        // scale * multiplicative mask; online softmax over t
        float p0[8], p1[8], tm[8];
#pragma unroll
        for (int i = 0; i < 8; ++i) {
            const int srow = stile * 16 + i + 8 * g;          // C/D row for elem i
            const float f0 = s0[i] * 0.125f * mask[(size_t)srow * S_ + tb + colx];
            const float f1 = s1[i] * 0.125f * mask[(size_t)srow * S_ + tb + 16 + colx];
            p0[i] = f0; p1[i] = f1;
            tm[i] = fmaxf(f0, f1);
        }
#pragma unroll
        for (int i = 0; i < 8; ++i) {
#pragma unroll
            for (int off = 1; off < 16; off <<= 1)            // 16-lane row groups
                tm[i] = fmaxf(tm[i], __shfl_xor(tm[i], off, 32));
        }
        float corr[8], rs[8];
#pragma unroll
        for (int i = 0; i < 8; ++i) {
            const float mn = fmaxf(mrun[i], tm[i]);
            corr[i] = __expf(mrun[i] - mn);
            mrun[i] = mn;
            p0[i] = __expf(p0[i] - mn);
            p1[i] = __expf(p1[i] - mn);
            rs[i] = p0[i] + p1[i];
        }
#pragma unroll
        for (int i = 0; i < 8; ++i) {
#pragma unroll
            for (int off = 1; off < 16; off <<= 1)
                rs[i] += __shfl_xor(rs[i], off, 32);
            lrun[i] = lrun[i] * corr[i] + rs[i];
        }
#pragma unroll
        for (int nt = 0; nt < 4; ++nt)
#pragma unroll
            for (int i = 0; i < 8; ++i)
                o[nt][i] *= corr[i];

        // C/D-layout P -> row-major 16x32 f16 in LDS -> A-layout operand
#pragma unroll
        for (int i = 0; i < 8; ++i) {
            const int r = i + 8 * g;
            pLDS[r * 32 + colx]      = (_Float16)p0[i];
            pLDS[r * 32 + colx + 16] = (_Float16)p1[i];
        }
        asm volatile("s_wait_dscnt 0" ::: "memory");          // DS in-order; fence scheduler too
        const v16h pa = ld_ab(&pLDS[colx * 32], g);

        // O += P (16x32) * V (32x64): 4 N-tiles from transposed V (contiguous along t)
#pragma unroll
        for (int nt = 0; nt < 4; ++nt) {
            const _Float16* vrow = Vt + (size_t)(nt * 16 + colx) * S_ + tb;
            o[nt] = wmma16(pa, ld_ab(vrow, g), o[nt]);
        }
    }

    // normalize and store to (s,b,d) f16 for the output projection
#pragma unroll
    for (int nt = 0; nt < 4; ++nt) {
#pragma unroll
        for (int i = 0; i < 8; ++i) {
            const int s = stile * 16 + i + 8 * g;
            const int d = h * 64 + nt * 16 + colx;
            attn[((size_t)(s * B_ + b)) * D_ + d] = (_Float16)(o[nt][i] / lrun[i]);
        }
    }
}

// ---------------------------------------------------------------- output projection
// attn (2048x1024,f16) @ Wo^T + bo -> out (f32); same 64Mx64N wave tiling.
__global__ __launch_bounds__(32) void out_gemm_kernel(
    const _Float16* __restrict__ attn, const _Float16* __restrict__ Wo16,
    const float* __restrict__ bo, float* __restrict__ out)
{
    const int lane   = threadIdx.x;
    const int colx   = lane & 15;
    const int g      = lane >> 4;
    const int mtile4 = blockIdx.x;  // 0..31
    const int n64    = blockIdx.y;  // 0..15

    const _Float16* arow[4];
#pragma unroll
    for (int mt = 0; mt < 4; ++mt)
        arow[mt] = attn + (size_t)(mtile4 * 64 + mt * 16 + colx) * D_;

    v8f acc[4][4];
#pragma unroll
    for (int mt = 0; mt < 4; ++mt)
#pragma unroll
        for (int nt = 0; nt < 4; ++nt) acc[mt][nt] = vzero8();

    for (int k = 0; k < D_; k += 32) {
        v16h a[4];
#pragma unroll
        for (int mt = 0; mt < 4; ++mt) a[mt] = ld_ab(arow[mt] + k, g);
#pragma unroll
        for (int nt = 0; nt < 4; ++nt) {
            const _Float16* wrow = Wo16 + (size_t)(n64 * 64 + nt * 16 + colx) * D_;
            v16h bm = ld_ab(wrow + k, g);
#pragma unroll
            for (int mt = 0; mt < 4; ++mt)
                acc[mt][nt] = wmma16(a[mt], bm, acc[mt][nt]);
        }
    }

#pragma unroll
    for (int nt = 0; nt < 4; ++nt) {
        const int   n  = n64 * 64 + nt * 16 + colx;
        const float bn = bo[n];
#pragma unroll
        for (int mt = 0; mt < 4; ++mt) {
#pragma unroll
            for (int i = 0; i < 8; ++i) {
                const int m = mtile4 * 64 + mt * 16 + i + 8 * g;
                out[(size_t)m * D_ + n] = acc[mt][nt][i] + bn;
            }
        }
    }
}

// ---------------------------------------------------------------- launch
extern "C" void kernel_launch(void* const* d_in, const int* in_sizes, int n_in,
                              void* d_out, int out_size, void* d_ws, size_t ws_size,
                              hipStream_t stream) {
    (void)in_sizes; (void)n_in; (void)out_size; (void)ws_size;

    const float* query = (const float*)d_in[0];
    const float* mask  = (const float*)d_in[1];
    const float* Wq    = (const float*)d_in[2];
    const float* bq    = (const float*)d_in[3];
    const float* Wk    = (const float*)d_in[4];
    const float* bk    = (const float*)d_in[5];
    const float* Wv    = (const float*)d_in[6];
    const float* bv    = (const float*)d_in[7];
    const float* Wo    = (const float*)d_in[8];
    const float* bo    = (const float*)d_in[9];
    float* out = (float*)d_out;

    char* ws = (char*)d_ws;
    const size_t MB = 1024 * 1024;
    _Float16* qf16 = (_Float16*)(ws);             // 2048x1024        (4 MB)
    _Float16* wf16 = (_Float16*)(ws + 4 * MB);    // 4x 1024x1024     (8 MB): q,k,v,o
    _Float16* xq   = (_Float16*)(ws + 12 * MB);   // (b,h,s,hd)       (4 MB)
    _Float16* xk   = (_Float16*)(ws + 16 * MB);   // (b,h,t,hd)       (4 MB)
    _Float16* xvT  = (_Float16*)(ws + 20 * MB);   // (b,h,hd,t)       (4 MB)
    _Float16* attn = (_Float16*)(ws + 24 * MB);   // (s,b,d)          (4 MB)

    const int NQ = M_ * D_;     // 2M
    const int NW = D_ * D_;     // 1M
    cvt_f16_kernel<<<(NQ + 255) / 256, 256, 0, stream>>>(query, qf16, NQ);
    cvt_f16_kernel<<<(NW + 255) / 256, 256, 0, stream>>>(Wq, wf16 + 0 * (size_t)NW, NW);
    cvt_f16_kernel<<<(NW + 255) / 256, 256, 0, stream>>>(Wk, wf16 + 1 * (size_t)NW, NW);
    cvt_f16_kernel<<<(NW + 255) / 256, 256, 0, stream>>>(Wv, wf16 + 2 * (size_t)NW, NW);
    cvt_f16_kernel<<<(NW + 255) / 256, 256, 0, stream>>>(Wo, wf16 + 3 * (size_t)NW, NW);

    qkv_gemm_kernel<<<dim3(M_ / 64, H_, 3), 32, 0, stream>>>(
        qf16, wf16, bq, bk, bv, xq, xk, xvT);

    attn_kernel<<<B_ * H_ * (S_ / 16), 32, 0, stream>>>(xq, xk, xvT, mask, attn);

    out_gemm_kernel<<<dim3(M_ / 64, D_ / 64), 32, 0, stream>>>(
        attn, wf16 + 3 * (size_t)NW, bo, out);
}
